// ATR_42013370089837
// MI455X (gfx1250) — matrix operand order repaired and checked by reading, not compiled, not run on hardware
//
#include <hip/hip_runtime.h>
#include <hip/hip_bf16.h>
#include <cstdint>

#define B_SZ 16
#define T_LEN 2048
#define D_INP 1024
#define D_HID 1024
#define NWG   16          // one workgroup per 64 hidden columns (cluster-sized)

typedef __attribute__((ext_vector_type(16))) __bf16 v16bf;
typedef __attribute__((ext_vector_type(8)))  __bf16 v8bf;
typedef __attribute__((ext_vector_type(8)))  float  v8f;
typedef __attribute__((ext_vector_type(4)))  unsigned int u32x4;
typedef __attribute__((ext_vector_type(8)))  int i32x8;
typedef __attribute__((ext_vector_type(4)))  int i32x4;

#if __has_builtin(__builtin_amdgcn_tensor_load_to_lds) && __has_builtin(__builtin_amdgcn_s_wait_tensorcnt)
#define ATR_HAVE_TDM 1
#else
#define ATR_HAVE_TDM 0
#endif

__device__ __forceinline__ v8f wmma_bf16(v16bf a, v16bf b, v8f c) {
  // D = A(16x32 bf16) * B(32x16 bf16) + C(16x16 f32)
  return __builtin_amdgcn_wmma_f32_16x16x32_bf16(false, a, false, b, (short)0, c,
                                                 false, false);
}

// ---------------------------------------------------------------------------
// fp32 -> bf16 weight conversion (Wi and Wh, 1M elements each)
// ---------------------------------------------------------------------------
__global__ __launch_bounds__(256)
void ATR_cvt(const float* __restrict__ Wi, const float* __restrict__ Wh,
             __bf16* __restrict__ Wib, __bf16* __restrict__ Whb) {
  const int i = blockIdx.x * 256 + threadIdx.x;
  Wib[i] = (__bf16)Wi[i];
  Whb[i] = (__bf16)Wh[i];
}

// ---------------------------------------------------------------------------
// Phase 1: P[b,t,n] = sum_i x[b,t,i] * Wi[n,i] + bi[n]   (stored into d_out)
// 256 threads = 8 waves; each wave owns one 16-row M tile, 16 accumulators.
// Software pipeline: LDS B-staging double-buffered (one sync/chunk, staging
// latency hidden under 16 WMMAs); B fragments double-buffered in registers
// (kills the load->wait->wmma serialization seen in round 2's disasm).
// ---------------------------------------------------------------------------
#define BROW 48                  // LDS row stride in elements (96B, bank padded)
#define BUFE (256 * BROW)        // elements per LDS staging buffer (24 KB)
__global__ __launch_bounds__(256)
void ATR_gemm_p(const float* __restrict__ x, const __bf16* __restrict__ Wib,
                const float* __restrict__ bi, float* __restrict__ P) {
  __shared__ __align__(32) __bf16 shB[2 * BUFE];   // 48 KB double buffer

  const int tid  = threadIdx.x;
  const int wave = tid >> 5;
  const int lane = tid & 31;
  const int mrow = lane & 15;   // A: M index / B,C: N index
  const int kh   = lane >> 4;   // K-half this lane holds
  const int m0   = (blockIdx.x * 8 + wave) << 4;
  const float* xrow = x + (size_t)(m0 + mrow) * D_INP;

  for (int ng = 0; ng < 4; ++ng) {         // 4 groups of 256 output columns
    const int nbase = ng << 8;
    v8f c[16];
#pragma unroll
    for (int nb = 0; nb < 16; ++nb)
      c[nb] = (v8f){0.f, 0.f, 0.f, 0.f, 0.f, 0.f, 0.f, 0.f};

    // ---- prologue: A(0) into regs, B(0) into LDS buffer 0
    float4 u0, u1, u2, u3;
    {
      const float* xp = xrow + kh * 16;
      u0 = *(const float4*)(xp);
      u1 = *(const float4*)(xp + 4);
      u2 = *(const float4*)(xp + 8);
      u3 = *(const float4*)(xp + 12);
      const __bf16* sp = Wib + (size_t)(nbase + tid) * D_INP;
      const v16bf s0 = *(const v16bf*)(sp);
      const v16bf s1 = *(const v16bf*)(sp + 16);
      *(v16bf*)(shB + tid * BROW)      = s0;
      *(v16bf*)(shB + tid * BROW + 16) = s1;
    }
    __syncthreads();

    for (int kc = 0; kc < 32; ++kc) {
      const bool more = (kc < 31);
      const int k1 = (kc + 1) << 5;
      // issue next chunk's A loads + B staging loads; latency hides under WMMAs
      float4 n0 = {}, n1 = {}, n2 = {}, n3 = {};
      v16bf s0 = {}, s1 = {};
      if (more) {
        const float* xp = xrow + k1 + kh * 16;
        n0 = *(const float4*)(xp);
        n1 = *(const float4*)(xp + 4);
        n2 = *(const float4*)(xp + 8);
        n3 = *(const float4*)(xp + 12);
        const __bf16* sp = Wib + (size_t)(nbase + tid) * D_INP + k1;
        s0 = *(const v16bf*)(sp);
        s1 = *(const v16bf*)(sp + 16);
      }
      // convert A(kc): lanes 0-15 K=k0+[0..16); hi lanes +16
      v16bf a;
      a[0]=(__bf16)u0.x;  a[1]=(__bf16)u0.y;  a[2]=(__bf16)u0.z;  a[3]=(__bf16)u0.w;
      a[4]=(__bf16)u1.x;  a[5]=(__bf16)u1.y;  a[6]=(__bf16)u1.z;  a[7]=(__bf16)u1.w;
      a[8]=(__bf16)u2.x;  a[9]=(__bf16)u2.y;  a[10]=(__bf16)u2.z; a[11]=(__bf16)u2.w;
      a[12]=(__bf16)u3.x; a[13]=(__bf16)u3.y; a[14]=(__bf16)u3.z; a[15]=(__bf16)u3.w;

      const __bf16* buf = shB + (kc & 1) * BUFE;
      v16bf bcur = *(const v16bf*)(buf + mrow * BROW + kh * 16);
#pragma unroll
      for (int nb = 0; nb < 16; ++nb) {     // B fragment double-buffered in regs
        v16bf bnxt = bcur;
        if (nb < 15)
          bnxt = *(const v16bf*)(buf + ((nb + 1) * 16 + mrow) * BROW + kh * 16);
        c[nb] = wmma_bf16(a, bcur, c[nb]);
        bcur = bnxt;
      }

      if (more) {                           // commit next staging buffer
        __bf16* nbuf = shB + ((kc + 1) & 1) * BUFE;
        *(v16bf*)(nbuf + tid * BROW)      = s0;
        *(v16bf*)(nbuf + tid * BROW + 16) = s1;
        u0 = n0; u1 = n1; u2 = n2; u3 = n3;
      }
      __syncthreads();
    }

#pragma unroll
    for (int nb = 0; nb < 16; ++nb) {
      const int col = nbase + nb * 16 + mrow;
      const float bias = bi[col];
#pragma unroll
      for (int r = 0; r < 8; ++r)           // C layout: VGPR r -> M = r (+8 hi lanes)
        P[(size_t)(m0 + r + kh * 8) * D_HID + col] = c[nb][r] + bias;
    }
  }
}

// ---------------------------------------------------------------------------
// Phase 2: persistent recurrent scan. 16 WGs x 128 threads (4 waves).
// WG g owns columns [g*64, g*64+64); Wh slice resident in LDS (padded
// half-row blocks, 2-way worst-case conflicts), preloaded via TDM + verified
// cooperative copy. Depth-2 software pipeline over the 32 K-chunks.
// ---------------------------------------------------------------------------
#define HR_BYTES 1040     // padded half-row block stride (512*2 + 16)
__global__ __launch_bounds__(128)
void ATR_scan(float* __restrict__ outP,       // (B,T,H) region of d_out; holds p, overwritten in place
              float* __restrict__ hFinal,     // d_out + B*T*H
              const int* __restrict__ lengths,
              const __bf16* __restrict__ Whb,
              const float* __restrict__ bh,
              float* __restrict__ h32,        // 2 * B*H fp32 (double buffer)
              __bf16* __restrict__ hbf,       // 2 * B*H bf16 (double buffer)
              unsigned* __restrict__ ctr) {
  __shared__ __align__(32) char shWh[128 * HR_BYTES];   // ~130 KB of 320 KB LDS

  const int g    = blockIdx.x;
  const int wave = threadIdx.x >> 5;
  const int lane = threadIdx.x & 31;
  const int nrow = lane & 15;
  const int kh   = lane >> 4;
  const int nloc = wave * 16 + nrow;        // row within the LDS Wh slice (0..63)
  const int n    = g * 64 + nloc;           // global hidden column

  // ---- async-tensor preload of the Wh slice (padded layout via D# pad fields) ----
#if ATR_HAVE_TDM
  if (threadIdx.x < 32) {                   // one wave issues the TDM descriptor
    const unsigned long long ga =
        (unsigned long long)(uintptr_t)(Whb + (size_t)g * 64 * D_HID);
    u32x4 g0;
    g0.x = 1u;                                           // count=1 (valid user D#)
    g0.y = 0u;                                           // lds_addr: shWh @ LDS offset 0
    g0.z = (unsigned)(ga & 0xFFFFFFFFull);               // global_addr[31:0]
    g0.w = (unsigned)((ga >> 32) & 0x01FFFFFFull) | (2u << 30);  // addr[56:32] | type=2
    i32x8 g1;
    // data_size=1 (2B) | pad_enable | pad_interval=7 (256 dwords) | pad_amount=3 (4 dwords)
    g1[0] = (int)0x07D10000;
    g1[1] = (int)(((unsigned)D_HID & 0xFFFFu) << 16);    // tensor_dim0 = 1024
    g1[2] = (int)(64u << 16);                            // tensor_dim1 = 64
    g1[3] = (int)((unsigned)D_HID << 16);                // tile_dim0 = 1024
    g1[4] = 64;                                          // tile_dim1 = 64, tile_dim2 = 0
    g1[5] = D_HID;                                       // tensor_dim0_stride = 1024
    g1[6] = 0;
    g1[7] = 0;
    i32x4 z4 = {};
#if defined(__clang_major__) && (__clang_major__ >= 23)
    i32x8 z8 = {};
    __builtin_amdgcn_tensor_load_to_lds(g0, g1, z4, z4, z8, 0);
#else
    __builtin_amdgcn_tensor_load_to_lds(g0, g1, z4, z4, 0);
#endif
    __builtin_amdgcn_s_wait_tensorcnt((short)0);
  }
  __syncthreads();
#endif
  // Authoritative cooperative copy into the same padded layout (also makes the
  // LDS writes visible to the compiler so ds_loads aren't folded to undef).
  {
    const __bf16* src = Whb + (size_t)g * 64 * D_HID;
    for (int i = (int)threadIdx.x; i < 128 * 64; i += (int)blockDim.x) {
      const int blk = i >> 6, c8 = i & 63;          // block 0..127, 8-elem chunk
      const int row = blk >> 1, half = blk & 1;
      const v8bf v = *(const v8bf*)(src + row * D_HID + half * 512 + c8 * 8);
      *(v8bf*)(shWh + blk * HR_BYTES + c8 * 16) = v;
    }
  }
  __syncthreads();

  const float bhn = bh[n];
  int lenb[8];
#pragma unroll
  for (int r = 0; r < 8; ++r) lenb[r] = lengths[r + kh * 8];

  for (int t = 0; t < T_LEN; ++t) {
    const int cur = t & 1, nxt = cur ^ 1;
    const __bf16* hb = hbf + (size_t)cur * (B_SZ * D_HID);
    const float*  hc = h32 + (size_t)cur * (B_SZ * D_HID);
    float*  hn32 = h32 + (size_t)nxt * (B_SZ * D_HID);
    __bf16* hnbf = hbf + (size_t)nxt * (B_SZ * D_HID);
    const __bf16* ha = hb + (size_t)(lane & 15) * D_HID;

    // hoist epilogue operands: their latency hides under the WMMA chain
    float pv[8], hpv[8];
#pragma unroll
    for (int r = 0; r < 8; ++r) {
      const int b = r + kh * 8;
      pv[r]  = outP[((size_t)b * T_LEN + t) * D_HID + n];
      hpv[r] = hc[b * D_HID + n];
    }

    auto load_a = [&](int kc) -> v16bf {
      const int k0 = kc << 5;
      const v8bf lo = *(const v8bf*)(ha + k0 + kh * 8);
      const v8bf hi = *(const v8bf*)(ha + k0 + 16 + kh * 8);
      v16bf a;
#pragma unroll
      for (int j = 0; j < 8; ++j) { a[j] = lo[j]; a[j + 8] = hi[j]; }
      return a;
    };
    auto load_b = [&](int kc) -> v16bf {
      const int k0 = kc << 5;
      const char* bp = shWh + (nloc * 2 + (k0 >> 9)) * HR_BYTES
                            + ((k0 & 511) + kh * 8) * 2;
      const v8bf b0 = *(const v8bf*)(bp);
      const v8bf b1 = *(const v8bf*)(bp + 32);
      v16bf b;
#pragma unroll
      for (int j = 0; j < 8; ++j) { b[j] = b0[j]; b[j + 8] = b1[j]; }
      return b;
    };

    // q tile = h_prev(16x1024) @ Wh_slice^T; depth-2 pipeline, 2 acc chains
    v8f acc0 = (v8f){0.f,0.f,0.f,0.f,0.f,0.f,0.f,0.f};
    v8f acc1 = (v8f){0.f,0.f,0.f,0.f,0.f,0.f,0.f,0.f};
    v16bf a_cur = load_a(0);
    v16bf b_cur = load_b(0);
#pragma unroll
    for (int kc = 0; kc < 32; ++kc) {
      v16bf a_nxt = a_cur, b_nxt = b_cur;
      if (kc < 31) { a_nxt = load_a(kc + 1); b_nxt = load_b(kc + 1); }
      if (kc & 1) acc1 = wmma_bf16(a_cur, b_cur, acc1);
      else        acc0 = wmma_bf16(a_cur, b_cur, acc0);
      a_cur = a_nxt; b_cur = b_nxt;
    }

    // gating epilogue; C layout: VGPR r -> batch b = r + kh*8, column n
#pragma unroll
    for (int r = 0; r < 8; ++r) {
      const int b = r + kh * 8;
      const size_t pidx = ((size_t)b * T_LEN + t) * D_HID + n;
      const float p  = pv[r];
      const float q  = acc0[r] + acc1[r] + bhn;
      const float hp = hpv[r];
      const float ig = 1.f / (1.f + __expf(-(p + q)));
      const float fg = 1.f / (1.f + __expf(-(p - q)));
      const float hnew = ig * p + fg * hp;
      const bool  m    = t < lenb[r];
      const float hnx  = m ? hnew : hp;
      outP[pidx]          = m ? hnew : 0.f;   // in-place: p already consumed
      hn32[b * D_HID + n] = hnx;
      hnbf[b * D_HID + n] = (__bf16)hnx;
      if (t + 1 < T_LEN)
        __builtin_prefetch(&outP[pidx + D_HID], 1, 0);   // next step's p row
    }

    // device-wide step barrier: release h writes, arrive, spin, acquire
    __syncthreads();
#if __has_builtin(__builtin_amdgcn_s_cluster_barrier)
    __builtin_amdgcn_s_cluster_barrier();   // NOP when not cluster-dispatched
#endif
    if (threadIdx.x == 0) {
      __threadfence();
      __hip_atomic_fetch_add(ctr, 1u, __ATOMIC_ACQ_REL, __HIP_MEMORY_SCOPE_AGENT);
      const unsigned tgt = (unsigned)(NWG * (t + 1));
      while (__hip_atomic_load(ctr, __ATOMIC_ACQUIRE, __HIP_MEMORY_SCOPE_AGENT) < tgt)
        __builtin_amdgcn_s_sleep(1);
    }
    __syncthreads();
  }

  // final hidden state lives in buffer 0 (2048 is even)
#pragma unroll
  for (int r = 0; r < 8; ++r) {
    const int b = r + kh * 8;
    hFinal[b * D_HID + n] = h32[b * D_HID + n];
  }
}

// ---------------------------------------------------------------------------
extern "C" void kernel_launch(void* const* d_in, const int* in_sizes, int n_in,
                              void* d_out, int out_size, void* d_ws, size_t ws_size,
                              hipStream_t stream) {
  const float* x       = (const float*)d_in[0];
  const int*   lengths = (const int*)d_in[1];
  const float* Wi      = (const float*)d_in[2];
  const float* bi      = (const float*)d_in[3];
  const float* Wh      = (const float*)d_in[4];
  const float* bh      = (const float*)d_in[5];

  float* outP   = (float*)d_out;
  float* hFinal = outP + (size_t)B_SZ * T_LEN * D_HID;

  char* ws = (char*)d_ws;
  __bf16*   Wib = (__bf16*)(ws);                       // 2 MB
  __bf16*   Whb = (__bf16*)(ws + (2u << 20));          // 2 MB
  float*    h32 = (float*) (ws + (4u << 20));          // 128 KB (2 buffers)
  __bf16*   hbf = (__bf16*)(ws + (4u << 20) + 2 * B_SZ * D_HID * sizeof(float)); // 64 KB
  unsigned* ctr = (unsigned*)(ws + (4u << 20) + 2 * B_SZ * D_HID * 6);

  // zero h double-buffers (h0 = 0) and the sync counter
  hipMemsetAsync(ws + (4u << 20), 0, (size_t)(2 * B_SZ * D_HID * 6 + 256), stream);

  ATR_cvt<<<dim3(D_HID * D_INP / 256), dim3(256), 0, stream>>>(Wi, Wh, Wib, Whb);
  ATR_gemm_p<<<dim3((B_SZ * T_LEN / 16) / 8), dim3(256), 0, stream>>>(x, Wib, bi, outP);
  ATR_scan<<<dim3(NWG), dim3(128), 0, stream>>>(outP, hFinal, lengths, Whb, bh,
                                                h32, hbf, ctr);
}
